// MultiAttributedModel_Iso_Merge_35124242546762
// MI455X (gfx1250) — matrix operand order repaired and checked by reading, not compiled, not run on hardware
//
#include <hip/hip_runtime.h>

typedef __attribute__((ext_vector_type(16))) __bf16 v16bf;
typedef __attribute__((ext_vector_type(8)))  float  v8f;

#define NN 50000
#define EE 800000

__device__ __forceinline__ unsigned short f32_to_bf16(float f) {
    union { float f; unsigned u; } x; x.f = f;
    unsigned u = x.u;
    u += 0x7FFFu + ((u >> 16) & 1u);   // round-to-nearest-even
    return (unsigned short)(u >> 16);
}

// ---------------------------------------------------------------------------
// Pack W [Din x DoutReal] f32 row-major into WMMA B-fragment bf16 layout.
// B 32x16 bf16 k order used here (wave32, same as A): lanes 0-15 hold col n, K = s*32 + 0..15 (16 vals);
// lanes 16-31 hold col n, K = s*32 + 16..31.  Stored per (colTile t, kstep s):
// out[((t*nSteps+s)*32 + lane)*16 + i]
// ---------------------------------------------------------------------------
__global__ void pack_w_kernel(const float* __restrict__ W, int Din, int DoutReal,
                              int nSteps, int colTiles, unsigned short* __restrict__ out) {
    int tid = blockIdx.x * blockDim.x + threadIdx.x;
    int total = colTiles * nSteps * 512;
    if (tid >= total) return;
    int i    = tid & 15;
    int lane = (tid >> 4) & 31;
    int ts   = tid >> 9;                 // t*nSteps + s
    int s    = ts % nSteps;
    int t    = ts / nSteps;
    int n    = t * 16 + (lane & 15);
    int k    = s * 32 + ((lane < 16) ? i : (16 + i));
    float v  = (n < DoutReal && k < Din) ? W[k * DoutReal + n] : 0.0f;
    out[tid] = f32_to_bf16(v);
}

// ---------------------------------------------------------------------------
// WMMA GEMM: out[m0..m0+15, :] = X[m0..m0+15, :Din] @ W  (bf16 x bf16 -> f32)
// blockDim = 32 * colTiles (one wave per 16-col tile), grid = N/16.
// Dynamic LDS: 16*Din bf16 staged X tile.
// ---------------------------------------------------------------------------
__global__ void gemm_wmma_kernel(const float* __restrict__ X,
                                 const unsigned short* __restrict__ Wp,
                                 float* __restrict__ out,
                                 int Din, int DoutReal, int ldOut, int nSteps) {
    extern __shared__ unsigned short sA[];      // [16][Din] bf16
    const int m0 = blockIdx.x * 16;

    // Stage 16 rows of X (contiguous 16*Din floats) as bf16. Loop count is
    // identical for all threads (16*Din divisible by blockDim) -> no divergence.
    const int total = 16 * Din;
    for (int idx = threadIdx.x; idx < total; idx += blockDim.x)
        sA[idx] = f32_to_bf16(X[(long)m0 * Din + idx]);
    __syncthreads();

    const int wave   = threadIdx.x >> 5;        // column tile
    const int lane   = threadIdx.x & 31;
    const int row    = lane & 15;
    const int halfHi = lane >> 4;

    v8f c = {0.f, 0.f, 0.f, 0.f, 0.f, 0.f, 0.f, 0.f};
    const uint4* WpV = reinterpret_cast<const uint4*>(Wp);

    for (int s = 0; s < nSteps; ++s) {
        // A frag 16x32 bf16: lane<16 -> K {0..15}; lane>=16 -> K {16..31}
        union { v16bf v; uint4 u[2]; } a;
        const unsigned short* abase = sA + row * Din + s * 32 + (halfHi ? 16 : 0);
        a.u[0] = *reinterpret_cast<const uint4*>(abase);
        a.u[1] = *reinterpret_cast<const uint4*>(abase + 8);
        // B frag from packed buffer: 32 contiguous bytes per lane
        union { v16bf v; uint4 u[2]; } b;
        int bi = ((wave * nSteps + s) * 32 + lane) * 2;   // in uint4 units
        b.u[0] = WpV[bi];
        b.u[1] = WpV[bi + 1];
        c = __builtin_amdgcn_wmma_f32_16x16x32_bf16(false, a.v, false, b.v,
                                                    (short)0, c, false, false);
    }

    // D layout: lane -> col (lane&15), VGPR r -> row r (lanes<16) / r+8 (lanes>=16)
    const int col = wave * 16 + row;
    if (col < DoutReal) {
        const int rbase = m0 + (halfHi ? 8 : 0);
        #pragma unroll
        for (int r = 0; r < 8; ++r)
            out[(long)(rbase + r) * ldOut + col] = c[r];
    }
}

// ---------------------------------------------------------------------------
// SpMM scatter: acc[dst[e], :] += w[e] * support[src[e], :]
// One thread per (edge, 4-float chunk); hardware f32 atomics (L2-resident).
// ---------------------------------------------------------------------------
__global__ void spmm_kernel(const float* __restrict__ sup, float* __restrict__ acc,
                            const int* __restrict__ src, const int* __restrict__ dst,
                            const float* __restrict__ w, int E, int F, int chunks) {
    int tid = blockIdx.x * blockDim.x + threadIdx.x;
    int total = E * chunks;
    if (tid >= total) return;
    int e = tid / chunks;
    int c = tid - e * chunks;
    int s = src[e], d = dst[e];
    float ww = w[e];
    const float4 v = *reinterpret_cast<const float4*>(sup + (long)s * F + c * 4);
    float* p = acc + (long)d * F + c * 4;
    unsafeAtomicAdd(p + 0, ww * v.x);
    unsafeAtomicAdd(p + 1, ww * v.y);
    unsafeAtomicAdd(p + 2, ww * v.z);
    unsafeAtomicAdd(p + 3, ww * v.w);
}

__global__ void zero_kernel(float4* __restrict__ p, int n4) {
    int i = blockIdx.x * blockDim.x + threadIdx.x;
    if (i < n4) p[i] = make_float4(0.f, 0.f, 0.f, 0.f);
}

__device__ __forceinline__ float lrelu(float x) { return x > 0.f ? x : 0.2f * x; }

__global__ void combine3_kernel(const float* __restrict__ a0, const float* __restrict__ a1,
                                const float* __restrict__ a2, float* __restrict__ z, int total) {
    int i = blockIdx.x * blockDim.x + threadIdx.x;
    if (i >= total) return;
    z[i] = (lrelu(a0[i]) + lrelu(a1[i]) + lrelu(a2[i])) * (1.0f / 3.0f);
}

// Student-t: V=1 -> exponent (V+1)/2 == 1, so q = 1/(1+dist), row-normalized.
__global__ void q_kernel(const float* __restrict__ z, const float* __restrict__ cl,
                         float* __restrict__ q, int N) {
    int n = blockIdx.x * blockDim.x + threadIdx.x;
    if (n >= N) return;
    float zr[20];
    #pragma unroll
    for (int d = 0; d < 20; ++d) zr[d] = z[(long)n * 20 + d];
    float qs[10], sum = 0.f;
    #pragma unroll
    for (int k = 0; k < 10; ++k) {
        float dist = 0.f;
        #pragma unroll
        for (int d = 0; d < 20; ++d) {
            float df = zr[d] - cl[k * 20 + d];
            dist += df * df;
        }
        float t = 1.f / (1.f + dist);
        qs[k] = t; sum += t;
    }
    float inv = 1.f / sum;
    #pragma unroll
    for (int k = 0; k < 10; ++k) q[(long)n * 10 + k] = qs[k] * inv;
}

extern "C" void kernel_launch(void* const* d_in, const int* in_sizes, int n_in,
                              void* d_out, int out_size, void* d_ws, size_t ws_size,
                              hipStream_t stream) {
    const int N = in_sizes[0] / 256;            // 50000
    const int E = in_sizes[3];                  // 800000

    // ---- workspace layout (bytes) ----
    char* ws = (char*)d_ws;
    unsigned short* packW = (unsigned short*)ws;                    // 3*43008 bf16
    float* support = (float*)(ws + 262144);                         // N*128 f32 (reused per branch)
    float* acc     = (float*)(ws + 262144 + (size_t)N * 128 * 4);   // 3 * N*128 f32
    float* z1      = (float*)(ws + 262144 + (size_t)N * 128 * 4 * 4);
    float* z2      = z1 + (size_t)N * 128;

    // per-layer config: {Din, DoutReal, nSteps (Din/32), colTiles (DoutP/16)}
    const int Din_[3]   = {256, 128, 64};
    const int Dout_[3]  = {128, 64, 20};
    const int nSt_[3]   = {8, 4, 2};
    const int cTil_[3]  = {8, 4, 2};
    const int packOff_[3] = {0, 32768, 40960};  // bf16 offsets within a branch's pack region
    const int packPerBranch = 43008;

    // ---- 1. pack all 9 weight matrices into WMMA B layout ----
    for (int br = 0; br < 3; ++br) {
        for (int l = 0; l < 3; ++l) {
            const float* W = (const float*)d_in[12 + br * 3 + l];
            int total = cTil_[l] * nSt_[l] * 512;
            pack_w_kernel<<<(total + 255) / 256, 256, 0, stream>>>(
                W, Din_[l], Dout_[l], nSt_[l], cTil_[l],
                packW + br * packPerBranch + packOff_[l]);
        }
    }

    // ---- 2. three GNN layers ----
    for (int l = 0; l < 3; ++l) {
        const int Din = Din_[l], F = Dout_[l];
        const int nSteps = nSt_[l], colTiles = cTil_[l];
        const int chunks = F / 4;               // 32, 16, 5

        // zero the 3 accumulators
        int n4 = 3 * N * F / 4;
        zero_kernel<<<(n4 + 255) / 256, 256, 0, stream>>>((float4*)acc, n4);

        for (int br = 0; br < 3; ++br) {
            const float* X = (l == 0) ? (const float*)d_in[br] : (l == 1 ? z1 : z2);
            gemm_wmma_kernel<<<N / 16, 32 * colTiles, 16 * Din * 2, stream>>>(
                X, packW + br * packPerBranch + packOff_[l], support, Din, F, F, nSteps);

            const int* srcp = (const int*)d_in[3 + 3 * br];
            const int* dstp = (const int*)d_in[4 + 3 * br];
            const float* wp = (const float*)d_in[5 + 3 * br];
            int totalS = E * chunks;
            spmm_kernel<<<(totalS + 255) / 256, 256, 0, stream>>>(
                support, acc + (size_t)br * N * F, srcp, dstp, wp, E, F, chunks);
        }

        float* zdst = (l == 0) ? z1 : (l == 1 ? z2 : (float*)d_out);
        int total = N * F;
        combine3_kernel<<<(total + 255) / 256, 256, 0, stream>>>(
            acc, acc + (size_t)N * F, acc + (size_t)2 * N * F, zdst, total);
    }

    // ---- 3. Student-t soft assignment into d_out[N*20 ..) ----
    q_kernel<<<(N + 255) / 256, 256, 0, stream>>>(
        (const float*)d_out, (const float*)d_in[21], (float*)d_out + (size_t)N * 20, N);
}
